// deform_up_20169166422163
// MI455X (gfx1250) — matrix operand order repaired and checked by reading, not compiled, not run on hardware
//
#include <hip/hip_runtime.h>

// ---------------------------------------------------------------------------
// Problem constants
// ---------------------------------------------------------------------------
#define Bsz     2
#define IN_CH   256
#define OUT_CH  128
#define Hh      96
#define Ww      96
#define HW      9216        // 96*96
#define KS      3
#define NOFF    18          // 2*KS*KS
#define CHUNK   2304        // pixels per im2col chunk (HW/4)
#define NCHUNK  4

typedef __attribute__((ext_vector_type(16))) __bf16   bf16x16;
typedef __attribute__((ext_vector_type(8)))  float    f32x8;
typedef __attribute__((ext_vector_type(4)))  unsigned u32x4;
typedef __attribute__((ext_vector_type(8)))  int      i32x8;
typedef __attribute__((ext_vector_type(4)))  int      i32x4;

union AFrag { bf16x16 v; uint4 u[2]; };

#define LDS_PITCH 40   // bf16 elements per n-row: 64B data + 16B pad (20-bank stride)

#if __has_builtin(__builtin_amdgcn_tensor_load_to_lds)
#define USE_TDM 1
#endif

// ---------------------------------------------------------------------------
// f32 -> bf16 convert (weights)
// ---------------------------------------------------------------------------
__global__ __launch_bounds__(256) void k_f2bf(const float* __restrict__ s,
                                              __bf16* __restrict__ d, int n) {
    int i = blockIdx.x * 256 + threadIdx.x;
    if (i < n) d[i] = (__bf16)s[i];
}

// ---------------------------------------------------------------------------
// im2col (COLUMN-major: col[n][K]) for halve conv, fused 2x bilinear upsample
// col[n*2304 + c*9 + a*3+b] = x2u[c, h+a-1, w+b-1] (zero pad), pixel=chunkBase+n
// ---------------------------------------------------------------------------
__global__ __launch_bounds__(256) void k_col_up(const float* __restrict__ x2b,
                                                __bf16* __restrict__ col,
                                                int chunkBase) {
    int idx = blockIdx.x * 256 + threadIdx.x;   // 9*CHUNK threads exactly
    int n = idx % CHUNK;
    int q = idx / CHUNK;                        // 0..8
    int a = q / 3, bb = q % 3;
    int p = chunkBase + n;
    int h = p / Ww, w = p % Ww;
    int i = h + a - 1, j = w + bb - 1;
    bool inb = ((unsigned)i < (unsigned)Hh) && ((unsigned)j < (unsigned)Ww);
    int y0 = 0, y1 = 0, x0 = 0, x1 = 0; float wy = 0.f, wx = 0.f;
    if (inb) {
        float sy = (float)i * (47.0f / 95.0f);
        float sx = (float)j * (47.0f / 95.0f);
        y0 = (int)floorf(sy); y1 = (y0 + 1 < 48) ? y0 + 1 : 47; wy = sy - (float)y0;
        x0 = (int)floorf(sx); x1 = (x0 + 1 < 48) ? x0 + 1 : 47; wx = sx - (float)x0;
    }
    __bf16* cp = col + (size_t)n * (IN_CH * 9) + q;
    for (int c = 0; c < IN_CH; ++c) {
        float v = 0.f;
        if (inb) {
            const float* xc = x2b + (size_t)c * 2304;   // 48*48
            float r0 = xc[y0 * 48 + x0] * (1.f - wx) + xc[y0 * 48 + x1] * wx;
            float r1 = xc[y1 * 48 + x0] * (1.f - wx) + xc[y1 * 48 + x1] * wx;
            v = r0 * (1.f - wy) + r1 * wy;
        }
        cp[(size_t)c * 9] = (__bf16)v;
    }
}

// ---------------------------------------------------------------------------
// Deformable im2col (COLUMN-major): col[n*(C*9) + c*9 + k]
// x: one batch [C,96,96]; off: one batch [96,96,18]
// ---------------------------------------------------------------------------
__global__ __launch_bounds__(256) void k_col_def(const float* __restrict__ x,
                                                 const float* __restrict__ off,
                                                 __bf16* __restrict__ col,
                                                 int C, int chunkBase) {
    int idx = blockIdx.x * 256 + threadIdx.x;   // 9*CHUNK threads exactly
    int n = idx % CHUNK;
    int k = idx / CHUNK;                        // 0..8
    int p = chunkBase + n;
    int h = p / Ww, w = p % Ww;
    const float* o = off + (size_t)p * NOFF;
    float px = o[k] + (float)(k / 3 - 1) + (float)(h + 1);
    float py = o[9 + k] + (float)(k % 3 - 1) + (float)(w + 1);
    float fx = floorf(px), fy = floorf(py);
    float xlt = fminf(fmaxf(fx, 0.f), 95.f);
    float ylt = fminf(fmaxf(fy, 0.f), 95.f);
    float xrb = fminf(fmaxf(fx + 1.f, 0.f), 95.f);
    float yrb = fminf(fmaxf(fy + 1.f, 0.f), 95.f);
    float pxc = fminf(fmaxf(px, 0.f), 95.f);
    float pyc = fminf(fmaxf(py, 0.f), 95.f);
    float glt = (1.f + (xlt - pxc)) * (1.f + (ylt - pyc));
    float grb = (1.f - (xrb - pxc)) * (1.f - (yrb - pyc));
    float glb = (1.f + (xlt - pxc)) * (1.f - (yrb - pyc));
    float grt = (1.f - (xrb - pxc)) * (1.f + (ylt - pyc));
    int ill = (int)xlt * Ww + (int)ylt;
    int irr = (int)xrb * Ww + (int)yrb;
    int ilr = (int)xlt * Ww + (int)yrb;
    int irl = (int)xrb * Ww + (int)ylt;
    __bf16* cp = col + (size_t)n * (C * 9) + k;
    for (int c = 0; c < C; ++c) {
        const float* xc = x + (size_t)c * HW;
        float v = glt * xc[ill] + grb * xc[irr] + glb * xc[ilr] + grt * xc[irl];
        cp[(size_t)c * 9] = (__bf16)v;
    }
}

// ---------------------------------------------------------------------------
// Offset conv: 3x3 pad=1, C -> 18, bias.  off layout: [b][96][96][18]
// ---------------------------------------------------------------------------
__global__ __launch_bounds__(256) void k_offconv(const float* __restrict__ x,
                                                 const float* __restrict__ W,
                                                 const float* __restrict__ bias,
                                                 float* __restrict__ off, int C) {
    int idx = blockIdx.x * 256 + threadIdx.x;   // Bsz*HW*18 exactly
    int b = idx / (HW * NOFF);
    int r = idx % (HW * NOFF);
    int p = r / NOFF, oc = r % NOFF;
    int h = p / Ww, w = p % Ww;
    const float* xb = x + (size_t)b * C * HW;
    float acc = bias[oc];
    for (int c = 0; c < C; ++c) {
        const float* xc = xb + (size_t)c * HW;
        const float* wc = W + ((size_t)oc * C + c) * 9;
#pragma unroll
        for (int dy = 0; dy < 3; ++dy) {
            int i = h + dy - 1;
            if ((unsigned)i >= (unsigned)Hh) continue;
#pragma unroll
            for (int dx = 0; dx < 3; ++dx) {
                int j = w + dx - 1;
                if ((unsigned)j >= (unsigned)Ww) continue;
                acc += wc[dy * 3 + dx] * xc[i * Ww + j];
            }
        }
    }
    off[(size_t)b * HW * NOFF + (size_t)p * NOFF + oc] = acc;
}

// ---------------------------------------------------------------------------
// TDM: load a 32(K,contig) x 64(N rows, stride K) bf16 tile into LDS with
// 16B padding after each 64B row -> LDS layout [n][LDS_PITCH=40].
// D# per CDNA5 ISA ch8: group0 {count,lds,global,type=2}, group1 {dims/strides}
// 6-arg builtin (this toolchain): (u32x4 g0, i32x8 g1, i32x4, i32x4, i32x8, cpol)
// ---------------------------------------------------------------------------
#ifdef USE_TDM
__device__ __forceinline__ void tdm_load_tile(const __bf16* gsrc,
                                              unsigned lds_byte_off, int K) {
    unsigned long long ga = (unsigned long long)(size_t)gsrc;
    u32x4 g0;
    g0.x = 1u;                                             // count=1, user desc
    g0.y = lds_byte_off;                                   // lds_addr
    g0.z = (unsigned)ga;                                   // global_addr lo
    g0.w = (unsigned)((ga >> 32) & 0x1FFFFFFu) | (2u << 30); // addr hi | type=2
    i32x8 g1;
    g1[0] = (int)((1u << 16)      // data_size = 1 (2 bytes)
                | (1u << 20)      // pad_enable
                | (3u << 22)      // pad_interval: 16 DWORDs (=64B) before pad
                | (3u << 25));    // pad_amount: 4 DWORDs (=16B)
    g1[1] = (int)(32u << 16);     // tensor_dim0 = 32 elements (contig K)
    g1[2] = (int)(64u << 16);     // tensor_dim1 = 64 rows
    g1[3] = (int)(32u << 16);     // tile_dim0 = 32
    g1[4] = (int)64;              // tile_dim1 = 64, tile_dim2 = 0
    g1[5] = (int)(unsigned)K;     // tensor_dim0_stride = K elements
    g1[6] = 0;
    g1[7] = 0;
    i32x4 z4 = { 0, 0, 0, 0 };
    i32x8 z8 = { 0, 0, 0, 0, 0, 0, 0, 0 };
    __builtin_amdgcn_tensor_load_to_lds(g0, g1, z4, z4, z8, 0);
}
#endif

// ---------------------------------------------------------------------------
// WMMA GEMM: D[128, CHUNK] = A[128,K](bf16) x colT[CHUNK,K](bf16, column-major)
// 4 waves; wave w owns M-tiles {w, w+4}; block owns a 64-wide N strip.
// Double-buffered LDS B tile, fed by TDM (or b128 copies as fallback).
// ---------------------------------------------------------------------------
__global__ __launch_bounds__(128) void k_gemm(const __bf16* __restrict__ A,
                                              const __bf16* __restrict__ col,
                                              float* __restrict__ outp,
                                              const float* __restrict__ bias,
                                              int K, int chunkBase, int relu) {
    __shared__ __bf16 ldsB[2][64 * LDS_PITCH];
    const int tid  = threadIdx.x;
    const int lane = tid & 31;
    const int wave = tid >> 5;              // 0..3
    const int nbase = blockIdx.x * 64;      // within chunk

    f32x8 acc0[4] = {};   // M-tile wave,    N subtiles 0..3
    f32x8 acc1[4] = {};   // M-tile wave+4

    // A fragment addressing (16-bit A 16x32 layout)
    const int arow  = wave * 16 + (lane & 15);
    const int akoff = (lane < 16) ? 0 : 8;
    const __bf16* aptr0 = A + (size_t)arow * K + akoff;
    const __bf16* aptr1 = aptr0 + (size_t)64 * K;

    // B fragment addressing (B 32x16: lanes 0-15 K=0..7,16..23, lanes 16-31 K=8..15,24..31)
    const int nloc  = lane & 15;
    const int bkoff = (lane < 16) ? 0 : 8;

    const __bf16* gtile = col + (size_t)nbase * K;   // 64 rows x K, row stride K
    const int nk = K >> 5;

    // ---- prologue: stage tile 0 into buffer 0 ----
#ifdef USE_TDM
    if (wave == 0) {
        tdm_load_tile(gtile, (unsigned)(size_t)&ldsB[0][0], K);
        __builtin_amdgcn_s_wait_tensorcnt(0);
    }
#else
#pragma unroll
    for (int r = 0; r < 2; ++r) {
        int idx = tid + r * 128;        // 256 x 16B pieces: 64 rows x 4
        int n = idx >> 2, part = idx & 3;
        *(uint4*)&ldsB[0][n * LDS_PITCH + part * 8] =
            *(const uint4*)(gtile + (size_t)n * K + part * 8);
    }
#endif
    __syncthreads();

    for (int kb = 0; kb < nk; ++kb) {
        const int cur = kb & 1;
        const bool more = (kb + 1 < nk);
#ifdef USE_TDM
        if (more && wave == 0)
            tdm_load_tile(gtile + (size_t)(kb + 1) * 32,
                          (unsigned)(size_t)&ldsB[cur ^ 1][0], K);
#else
        uint4 stg[2];
        if (more) {
#pragma unroll
            for (int r = 0; r < 2; ++r) {
                int idx = tid + r * 128;
                int n = idx >> 2, part = idx & 3;
                stg[r] = *(const uint4*)(gtile + (size_t)n * K + (kb + 1) * 32 + part * 8);
            }
        }
#endif
        AFrag a0, a1;
        a0.u[0] = *(const uint4*)(aptr0 + kb * 32);
        a0.u[1] = *(const uint4*)(aptr0 + kb * 32 + 16);
        a1.u[0] = *(const uint4*)(aptr1 + kb * 32);
        a1.u[1] = *(const uint4*)(aptr1 + kb * 32 + 16);

        const __bf16* bbase = &ldsB[cur][nloc * LDS_PITCH + bkoff];
#pragma unroll
        for (int ns = 0; ns < 4; ++ns) {
            AFrag bf;
            const __bf16* bp = bbase + ns * 16 * LDS_PITCH;
            bf.u[0] = *(const uint4*)bp;
            bf.u[1] = *(const uint4*)(bp + 16);
            acc0[ns] = __builtin_amdgcn_wmma_f32_16x16x32_bf16(
                false, a0.v, false, bf.v, (short)0, acc0[ns], false, false);
            acc1[ns] = __builtin_amdgcn_wmma_f32_16x16x32_bf16(
                false, a1.v, false, bf.v, (short)0, acc1[ns], false, false);
        }
#ifdef USE_TDM
        if (more && wave == 0)
            __builtin_amdgcn_s_wait_tensorcnt(0);
#else
        if (more) {
#pragma unroll
            for (int r = 0; r < 2; ++r) {
                int idx = tid + r * 128;
                int n = idx >> 2, part = idx & 3;
                *(uint4*)&ldsB[cur ^ 1][n * LDS_PITCH + part * 8] = stg[r];
            }
        }
#endif
        __syncthreads();
    }

    // ---- epilogue: C/D 16x16 f32 layout (VGPR r: lanes0-15 M=r, 16-31 M=r+8)
    const int m0 = wave * 16 + ((lane < 16) ? 0 : 8);
#pragma unroll
    for (int mt = 0; mt < 2; ++mt) {
        const int mbase = m0 + mt * 64;
        float4 bv0 = { 0.f, 0.f, 0.f, 0.f }, bv1 = { 0.f, 0.f, 0.f, 0.f };
        if (bias) {
            bv0 = *(const float4*)&bias[mbase];
            bv1 = *(const float4*)&bias[mbase + 4];
        }
#pragma unroll
        for (int ns = 0; ns < 4; ++ns) {
            f32x8 acc = mt == 0 ? acc0[ns] : acc1[ns];
            const int ncol = chunkBase + nbase + ns * 16 + nloc;
#pragma unroll
            for (int r = 0; r < 8; ++r) {
                float v = acc[r] + (r < 4 ? (&bv0.x)[r] : (&bv1.x)[r - 4]);
                if (relu) v = fmaxf(v, 0.f);
                outp[(size_t)(mbase + r) * HW + ncol] = v;
            }
        }
    }
}

// ---------------------------------------------------------------------------
// BatchNorm (training-mode batch stats) : one block per channel
// ---------------------------------------------------------------------------
__global__ __launch_bounds__(256) void k_bnstats(const float* __restrict__ y,
                                                 float* __restrict__ stats) {
    __shared__ float ss[256], sq[256];
    int c = blockIdx.x, tid = threadIdx.x;
    float s = 0.f, q = 0.f;
    for (int e = tid; e < Bsz * HW; e += 256) {
        int b = e / HW, i = e % HW;
        float v = y[((size_t)b * OUT_CH + c) * HW + i];
        s += v; q += v * v;
    }
    ss[tid] = s; sq[tid] = q;
    __syncthreads();
    for (int st = 128; st > 0; st >>= 1) {
        if (tid < st) { ss[tid] += ss[tid + st]; sq[tid] += sq[tid + st]; }
        __syncthreads();
    }
    if (tid == 0) {
        float m = ss[0] / (float)(Bsz * HW);
        stats[c] = m;
        stats[OUT_CH + c] = sq[0] / (float)(Bsz * HW) - m * m;
    }
}

__global__ __launch_bounds__(256) void k_bnapply(float* __restrict__ y,
                                                 const float* __restrict__ stats,
                                                 const float* __restrict__ g,
                                                 const float* __restrict__ be) {
    int idx = blockIdx.x * 256 + threadIdx.x;   // Bsz*OUT_CH*HW exactly
    int c = (idx / HW) % OUT_CH;
    float m = stats[c];
    float v = stats[OUT_CH + c];
    float r = rsqrtf(v + 1e-5f);
    float val = (y[idx] - m) * r * g[c] + be[c];
    y[idx] = fmaxf(val, 0.f);
}

// ---------------------------------------------------------------------------
// Host orchestration
// ---------------------------------------------------------------------------
extern "C" void kernel_launch(void* const* d_in, const int* in_sizes, int n_in,
                              void* d_out, int out_size, void* d_ws, size_t ws_size,
                              hipStream_t stream) {
    (void)in_sizes; (void)n_in; (void)out_size; (void)ws_size;
    const float* x1   = (const float*)d_in[0];
    const float* x2   = (const float*)d_in[1];
    const float* chW  = (const float*)d_in[2];
    const float* chB  = (const float*)d_in[3];
    const float* p1W  = (const float*)d_in[4];
    const float* p1B  = (const float*)d_in[5];
    const float* c1W  = (const float*)d_in[6];
    const float* bn1g = (const float*)d_in[7];
    const float* bn1b = (const float*)d_in[8];
    const float* p2W  = (const float*)d_in[9];
    const float* p2B  = (const float*)d_in[10];
    const float* c2W  = (const float*)d_in[11];
    const float* bn2g = (const float*)d_in[12];
    const float* bn2b = (const float*)d_in[13];
    float* out = (float*)d_out;

    // workspace layout (all offsets 256B aligned)
    char* w = (char*)d_ws;
    float*  xcat  = (float*)(w + 0);                       // [2,256,96,96]   18,874,368 B
    float*  offb  = (float*)(w + 18874368);                // [2,96,96,18]     1,327,104 B
    float*  y1    = (float*)(w + 20201472);                // [2,128,96,96]    9,437,184 B
    __bf16* wch   = (__bf16*)(w + 29638656);               // [128,2304]         589,824 B
    __bf16* wc1   = (__bf16*)(w + 30228480);               // [128,2304]         589,824 B
    __bf16* wc2   = (__bf16*)(w + 30818304);               // [128,1152]         294,912 B
    __bf16* col   = (__bf16*)(w + 31113216);               // colT [2304,2304] 10,616,832 B
    float*  stats = (float*)(w + 41730048);                // [256]

    // weights -> bf16
    k_f2bf<<<294912 / 256, 256, 0, stream>>>(chW, wch, 294912);
    k_f2bf<<<294912 / 256, 256, 0, stream>>>(c1W, wc1, 294912);
    k_f2bf<<<147456 / 256, 256, 0, stream>>>(c2W, wc2, 147456);

    // x1 -> second half of concat
    for (int b = 0; b < Bsz; ++b)
        (void)hipMemcpyAsync(xcat + ((size_t)b * IN_CH + OUT_CH) * HW,
                             x1 + (size_t)b * OUT_CH * HW,
                             (size_t)OUT_CH * HW * sizeof(float),
                             hipMemcpyDeviceToDevice, stream);

    const int colBlocks  = 9 * CHUNK / 256;  // 81
    const int gemmBlocks = CHUNK / 64;       // 36

    // halve conv (upsample fused into im2col) -> first half of concat
    for (int b = 0; b < Bsz; ++b)
        for (int ch = 0; ch < NCHUNK; ++ch) {
            k_col_up<<<colBlocks, 256, 0, stream>>>(x2 + (size_t)b * IN_CH * 2304,
                                                    col, ch * CHUNK);
            k_gemm<<<gemmBlocks, 128, 0, stream>>>(wch, col,
                                                   xcat + (size_t)b * IN_CH * HW,
                                                   chB, 2304, ch * CHUNK, 1);
        }

    // deform conv 1
    k_offconv<<<Bsz * HW * NOFF / 256, 256, 0, stream>>>(xcat, p1W, p1B, offb, IN_CH);
    for (int b = 0; b < Bsz; ++b)
        for (int ch = 0; ch < NCHUNK; ++ch) {
            k_col_def<<<colBlocks, 256, 0, stream>>>(xcat + (size_t)b * IN_CH * HW,
                                                     offb + (size_t)b * HW * NOFF,
                                                     col, IN_CH, ch * CHUNK);
            k_gemm<<<gemmBlocks, 128, 0, stream>>>(wc1, col,
                                                   y1 + (size_t)b * OUT_CH * HW,
                                                   nullptr, 2304, ch * CHUNK, 0);
        }
    k_bnstats<<<OUT_CH, 256, 0, stream>>>(y1, stats);
    k_bnapply<<<Bsz * OUT_CH * HW / 256, 256, 0, stream>>>(y1, stats, bn1g, bn1b);

    // deform conv 2
    k_offconv<<<Bsz * HW * NOFF / 256, 256, 0, stream>>>(y1, p2W, p2B, offb, OUT_CH);
    for (int b = 0; b < Bsz; ++b)
        for (int ch = 0; ch < NCHUNK; ++ch) {
            k_col_def<<<colBlocks, 256, 0, stream>>>(y1 + (size_t)b * OUT_CH * HW,
                                                     offb + (size_t)b * HW * NOFF,
                                                     col, OUT_CH, ch * CHUNK);
            k_gemm<<<gemmBlocks, 128, 0, stream>>>(wc2, col,
                                                   out + (size_t)b * OUT_CH * HW,
                                                   nullptr, 1152, ch * CHUNK, 0);
        }
    k_bnstats<<<OUT_CH, 256, 0, stream>>>(out, stats);
    k_bnapply<<<Bsz * OUT_CH * HW / 256, 256, 0, stream>>>(out, stats, bn2g, bn2b);
}
